// ModernBERTFlashAttnBlock_BMHd_85529978733230
// MI455X (gfx1250) — compile-verified, matrix operand order checked by reading, not run on hardware
//
#include <hip/hip_runtime.h>
#include <hip/hip_bf16.h>

// ModernBERT block on gfx1250 (MI455X), compile-only target.
// All matmuls through v_wmma_f32_16x16x32_bf16 (f32 accumulate); flash
// attention with online softmax; double-buffered LDS GEMM pipeline with
// gfx1250 async global->LDS loads tracked by ASYNCcnt.

#define B_  2
#define M_  2048
#define D_  1024
#define H_  16
#define DH_ 64
#define FF_ 2624
#define MT_TOT (B_ * M_)   // 4096 rows

typedef __attribute__((ext_vector_type(16))) __bf16 v16bf;
typedef __attribute__((ext_vector_type(8)))  __bf16 v8bf;
typedef __attribute__((ext_vector_type(8)))  float  v8f;
typedef __attribute__((ext_vector_type(4)))  int    v4i;

#if defined(__AMDGCN__) && __has_builtin(__builtin_amdgcn_global_load_async_to_lds_b128)
#define HAVE_ASYNC_LDS 1
#else
#define HAVE_ASYNC_LDS 0
#endif

#if HAVE_ASYNC_LDS
__device__ __forceinline__ void async_ld_b128(const __bf16* g, __bf16* l) {
    __builtin_amdgcn_global_load_async_to_lds_b128(
        (v4i*)const_cast<__bf16*>(g), (v4i*)l, /*offset=*/0, /*cpol=*/0);
}
__device__ __forceinline__ void wait_async0() {
#if __has_builtin(__builtin_amdgcn_s_wait_asynccnt)
    __builtin_amdgcn_s_wait_asynccnt(0);
#else
    asm volatile("s_wait_asynccnt 0x0" ::: "memory");
#endif
}
#endif

__device__ __forceinline__ v8f wmma_bf16(v16bf a, v16bf b, v8f c) {
    return __builtin_amdgcn_wmma_f32_16x16x32_bf16(
        /*neg_a=*/false, a, /*neg_b=*/false, b,
        /*c_mod=*/(short)0, c, /*reuse_a=*/false, /*reuse_b=*/false);
}

// Load a 16x32 bf16 operand fragment (A layout; B uses the same per-lane
// pattern with lane = output column).  CDNA5 16-bit layout:
//   lanes 0-15: M=lane,    K in [0,8) and [16,24)
//   lanes16-31: M=lane-16, K in [8,16) and [24,32)
__device__ __forceinline__ v16bf load_frag(const __bf16* base, int ld, int lane) {
    int r  = lane & 15;
    int k0 = (lane < 16) ? 0 : 8;
    const __bf16* p = base + (size_t)r * ld + k0;
    v8bf lo = *(const v8bf*)(p);
    v8bf hi = *(const v8bf*)(p + 16);
    return __builtin_shufflevector(lo, hi, 0,1,2,3,4,5,6,7,8,9,10,11,12,13,14,15);
}

// ---------------------------------------------------------------------------
// f32 -> bf16 pack (weights)
// ---------------------------------------------------------------------------
__global__ __launch_bounds__(256) void pack_bf16_kernel(
    const float* __restrict__ src, __bf16* __restrict__ dst, int n) {
    int i = blockIdx.x * 256 + threadIdx.x;
    if (i < n) dst[i] = (__bf16)src[i];
}

// ---------------------------------------------------------------------------
// LayerNorm (f32 in) -> bf16 out.  One block per row, 256 threads.
// ---------------------------------------------------------------------------
__global__ __launch_bounds__(256) void layernorm_bf16_kernel(
    const float* __restrict__ x, const float* __restrict__ w,
    const float* __restrict__ b, __bf16* __restrict__ out, int Dd) {
    int row = blockIdx.x;
    const float* xr = x + (size_t)row * Dd;
    __shared__ float redS[8], redSS[8];
    __shared__ float mu_s, rstd_s;
    float s = 0.f, ss = 0.f;
    for (int i = threadIdx.x; i < Dd; i += 256) {
        float v = xr[i]; s += v; ss += v * v;
    }
    for (int off = 16; off; off >>= 1) {
        s  += __shfl_down(s, off, 32);
        ss += __shfl_down(ss, off, 32);
    }
    if ((threadIdx.x & 31) == 0) { redS[threadIdx.x >> 5] = s; redSS[threadIdx.x >> 5] = ss; }
    __syncthreads();
    if (threadIdx.x == 0) {
        float S = 0.f, SS = 0.f;
        for (int i = 0; i < 8; i++) { S += redS[i]; SS += redSS[i]; }
        float mu  = S / Dd;
        float var = SS / Dd - mu * mu;
        mu_s = mu; rstd_s = rsqrtf(var + 1e-5f);
    }
    __syncthreads();
    float mu = mu_s, rs = rstd_s;
    for (int i = threadIdx.x; i < Dd; i += 256)
        out[(size_t)row * Dd + i] = (__bf16)((xr[i] - mu) * rs * w[i] + b[i]);
}

// ---------------------------------------------------------------------------
// Generic bf16 GEMM: C[M,N] = A[M,K] @ Bw[N,K]^T (+bias) (+resid), f32 out.
// Block tile 128x128, 8 waves (4 in M x 2 in N), wave tile 32x64,
// double-buffered LDS, K step 32.  8 WMMAs per wave per K step.
// ---------------------------------------------------------------------------
#define GMT 128
#define GNT 128
#define GKT 32
#define GKP (GKT + 8)   // LDS pad (16B) to dodge bank conflicts

__global__ __launch_bounds__(256) void gemm_bf16_kernel(
    const __bf16* __restrict__ A, const __bf16* __restrict__ Bw,
    const float* __restrict__ bias, const float* __restrict__ resid,
    float* __restrict__ C, int Mtot, int N, int K) {
    __shared__ alignas(16) __bf16 As[2][GMT][GKP];
    __shared__ alignas(16) __bf16 Bs[2][GNT][GKP];

    int tid  = threadIdx.x;
    int lane = tid & 31;
    int wave = tid >> 5;
    int m0 = blockIdx.y * GMT;
    int n0 = blockIdx.x * GNT;
    int wm = (wave & 3) * 32;   // wave row offset inside tile
    int wn = (wave >> 2) * 64;  // wave col offset inside tile

    v8f acc[2][4] = {};

    // Staging: 256 threads cover a 128x32 tile for A and for B.
    int sr = tid >> 1;          // 0..127 : staging row
    int sk = (tid & 1) * 16;    // 0/16   : staging col

    const __bf16* agp = A  + (size_t)(m0 + sr) * K + sk;
    const __bf16* bgp = Bw + (size_t)(n0 + sr) * K + sk;
    int nstage = K / GKT;

    auto compute = [&](int buf) {
        v16bf af0 = load_frag(&As[buf][wm][0],      GKP, lane);
        v16bf af1 = load_frag(&As[buf][wm + 16][0], GKP, lane);
        for (int j = 0; j < 4; ++j) {
            v16bf bfj = load_frag(&Bs[buf][wn + j * 16][0], GKP, lane);
            acc[0][j] = wmma_bf16(af0, bfj, acc[0][j]);
            acc[1][j] = wmma_bf16(af1, bfj, acc[1][j]);
        }
    };

#if HAVE_ASYNC_LDS
    auto issue_stage = [&](int s, int buf) {
        const __bf16* ag = agp + s * GKT;
        const __bf16* bg = bgp + s * GKT;
        async_ld_b128(ag,     &As[buf][sr][sk]);
        async_ld_b128(ag + 8, &As[buf][sr][sk + 8]);
        async_ld_b128(bg,     &Bs[buf][sr][sk]);
        async_ld_b128(bg + 8, &Bs[buf][sr][sk + 8]);
    };
    issue_stage(0, 0);
    wait_async0();
    __syncthreads();
    for (int s = 0; s < nstage; ++s) {
        int buf = s & 1;
        if (s + 1 < nstage) issue_stage(s + 1, buf ^ 1);
        compute(buf);
        wait_async0();
        __syncthreads();
    }
#else
    v8bf ra0, ra1, rb0, rb1;
    auto gload = [&](int s) {
        const __bf16* ag = agp + s * GKT;
        const __bf16* bg = bgp + s * GKT;
        ra0 = *(const v8bf*)(ag); ra1 = *(const v8bf*)(ag + 8);
        rb0 = *(const v8bf*)(bg); rb1 = *(const v8bf*)(bg + 8);
    };
    auto lstore = [&](int buf) {
        *(v8bf*)&As[buf][sr][sk]     = ra0;
        *(v8bf*)&As[buf][sr][sk + 8] = ra1;
        *(v8bf*)&Bs[buf][sr][sk]     = rb0;
        *(v8bf*)&Bs[buf][sr][sk + 8] = rb1;
    };
    gload(0);
    lstore(0);
    __syncthreads();
    for (int s = 0; s < nstage; ++s) {
        int buf = s & 1;
        if (s + 1 < nstage) gload(s + 1);   // issue global loads early
        compute(buf);                        // overlap with loads in flight
        if (s + 1 < nstage) lstore(buf ^ 1);
        __syncthreads();
    }
#endif

    // Epilogue.  C layout: VGPR r, lane l -> row = r + 8*(l>=16), col = l&15.
    int halfo = (lane >> 4) << 3;
    for (int i = 0; i < 2; i++) {
        for (int j = 0; j < 4; j++) {
            int col = n0 + wn + j * 16 + (lane & 15);
            float bv = bias ? bias[col] : 0.f;
            for (int r = 0; r < 8; r++) {
                int row = m0 + wm + i * 16 + r + halfo;
                float v = acc[i][j][r] + bv;
                size_t idx = (size_t)row * N + col;
                if (resid) v += resid[idx];
                C[idx] = v;
            }
        }
    }
}

// ---------------------------------------------------------------------------
// RoPE + bf16 pack.  qkv f32 [B*M, 3*D] -> q,k bf16 [B,H,M,DH], v^T bf16
// [B,H,DH,M] (transposed so attention's P@V B-fragments load contiguously).
// Grid = B*M rows, block = H*32 = 512 threads.
// ---------------------------------------------------------------------------
__global__ __launch_bounds__(512) void rope_pack_kernel(
    const float* __restrict__ qkv, __bf16* __restrict__ q,
    __bf16* __restrict__ k, __bf16* __restrict__ vT) {
    int row = blockIdx.x;          // b*M + m
    int b = row / M_;
    int m = row - b * M_;
    int h = threadIdx.x >> 5;
    int t = threadIdx.x & 31;      // half-dim index (0..31)

    const float* base = qkv + (size_t)row * (3 * D_);
    float inv = __powf(10000.f, -(2.f * t) / 64.f);
    float fr  = (float)m * inv;
    float c   = __cosf(fr);
    float sn  = __sinf(fr);
    int off   = h * DH_;

    float q1 = base[0 * D_ + off + t], q2 = base[0 * D_ + off + t + 32];
    float k1 = base[1 * D_ + off + t], k2 = base[1 * D_ + off + t + 32];
    float v1 = base[2 * D_ + off + t], v2 = base[2 * D_ + off + t + 32];

    size_t qo = ((size_t)((b * H_ + h) * M_) + m) * DH_;
    q[qo + t]      = (__bf16)(q1 * c - q2 * sn);
    q[qo + t + 32] = (__bf16)(q1 * sn + q2 * c);
    k[qo + t]      = (__bf16)(k1 * c - k2 * sn);
    k[qo + t + 32] = (__bf16)(k1 * sn + k2 * c);

    size_t vo = (size_t)(b * H_ + h) * DH_ * M_;
    vT[vo + (size_t)t * M_ + m]        = (__bf16)v1;
    vT[vo + (size_t)(t + 32) * M_ + m] = (__bf16)v2;
}

// ---------------------------------------------------------------------------
// Flash attention.  Grid (M/128, H, B); 8 waves/block, 16 query rows/wave.
// S = Q(16x64) @ K^T via 2 chained WMMAs; online softmax; P re-laid out
// through per-wave LDS into an A-fragment; O += P @ V via 4 WMMAs.
// Output written bf16 into [B*M, D] with heads interleaved (h*64+d).
// ---------------------------------------------------------------------------
__global__ __launch_bounds__(256) void flash_attn_kernel(
    const __bf16* __restrict__ Q, const __bf16* __restrict__ Kb,
    const __bf16* __restrict__ Vt, const int* __restrict__ mask,
    __bf16* __restrict__ O) {
    __shared__ alignas(16) __bf16 plds[8][16][GKP];

    int lane = threadIdx.x & 31;
    int wave = threadIdx.x >> 5;
    int b = blockIdx.z, h = blockIdx.y;
    int mbase = blockIdx.x * 128 + wave * 16;
    int halfo = (lane >> 4) << 3;

    const __bf16* qp = Q + ((size_t)((b * H_ + h) * M_) + mbase) * DH_;
    v16bf qf0 = load_frag(qp,      DH_, lane);
    v16bf qf1 = load_frag(qp + 32, DH_, lane);

    const __bf16* kbase = Kb + (size_t)((b * H_ + h) * M_) * DH_;
    const __bf16* vbase = Vt + (size_t)(b * H_ + h) * DH_ * M_;
    const int*    mk    = mask + (size_t)b * M_;

    v8f oacc[4] = {};
    float mrow[8], lrow[8];
    for (int r = 0; r < 8; r++) { mrow[r] = -INFINITY; lrow[r] = 0.f; }
    const float scale = 0.125f;   // 1/sqrt(64)

    for (int n0 = 0; n0 < M_; n0 += 32) {
        v8f s[2] = {};
        for (int j = 0; j < 2; j++) {
            const __bf16* kp = kbase + (size_t)(n0 + j * 16) * DH_;
            v16bf kf0 = load_frag(kp,      DH_, lane);
            v16bf kf1 = load_frag(kp + 32, DH_, lane);
            s[j] = wmma_bf16(qf0, kf0, s[j]);
            s[j] = wmma_bf16(qf1, kf1, s[j]);
        }
        // scale + mask (column validity per lane)
        float sc[2][8];
        for (int j = 0; j < 2; j++) {
            int ncol = n0 + j * 16 + (lane & 15);
            bool keep = mk[ncol] > 0;
            for (int r = 0; r < 8; r++)
                sc[j][r] = keep ? s[j][r] * scale : -INFINITY;
        }
        // online softmax per owned row (rows live in 16-lane halves)
        for (int r = 0; r < 8; r++) {
            float tmax = fmaxf(sc[0][r], sc[1][r]);
            tmax = fmaxf(tmax, __shfl_xor(tmax, 1, 32));
            tmax = fmaxf(tmax, __shfl_xor(tmax, 2, 32));
            tmax = fmaxf(tmax, __shfl_xor(tmax, 4, 32));
            tmax = fmaxf(tmax, __shfl_xor(tmax, 8, 32));
            float mn = fmaxf(mrow[r], tmax);
            float alpha, p0, p1;
            if (mn == -INFINITY) { alpha = 1.f; p0 = 0.f; p1 = 0.f; }
            else {
                alpha = __expf(mrow[r] - mn);   // exp(-inf)=0 on first hit
                p0 = (sc[0][r] == -INFINITY) ? 0.f : __expf(sc[0][r] - mn);
                p1 = (sc[1][r] == -INFINITY) ? 0.f : __expf(sc[1][r] - mn);
            }
            float psum = p0 + p1;
            psum += __shfl_xor(psum, 1, 32);
            psum += __shfl_xor(psum, 2, 32);
            psum += __shfl_xor(psum, 4, 32);
            psum += __shfl_xor(psum, 8, 32);
            lrow[r] = lrow[r] * alpha + psum;
            mrow[r] = mn;
            for (int d = 0; d < 4; d++) oacc[d][r] *= alpha;
            // stash P (bf16) in per-wave LDS in C layout for re-read as A-frag
            int prow = r + halfo;
            plds[wave][prow][lane & 15]        = (__bf16)p0;
            plds[wave][prow][(lane & 15) + 16] = (__bf16)p1;
        }
        // P (16x32) as A-fragment, V^T rows as B-fragments, K-dim = 32 keys
        v16bf pf = load_frag(&plds[wave][0][0], GKP, lane);
        for (int d = 0; d < 4; d++) {
            v16bf vf = load_frag(vbase + (size_t)(d * 16) * M_ + n0, M_, lane);
            oacc[d] = wmma_bf16(pf, vf, oacc[d]);
        }
    }

    for (int d = 0; d < 4; d++) {
        int col = h * DH_ + d * 16 + (lane & 15);
        for (int r = 0; r < 8; r++) {
            int row = mbase + r + halfo;
            float denom = lrow[r];
            float v = (denom > 0.f) ? oacc[d][r] / denom : 0.f;
            O[(size_t)(b * M_ + row) * D_ + col] = (__bf16)v;
        }
    }
}

// ---------------------------------------------------------------------------
// GEGLU: act = gelu_exact(g[:, :FF]) * g[:, FF:], bf16 out [B*M, FF]
// ---------------------------------------------------------------------------
__global__ __launch_bounds__(256) void geglu_kernel(
    const float* __restrict__ g, __bf16* __restrict__ out) {
    size_t i = (size_t)blockIdx.x * 256 + threadIdx.x;
    size_t n = (size_t)MT_TOT * FF_;
    if (i >= n) return;
    size_t row = i / FF_;
    size_t f   = i - row * FF_;
    float a  = g[row * (2 * FF_) + f];
    float gt = g[row * (2 * FF_) + FF_ + f];
    float ge = 0.5f * a * (1.f + erff(a * 0.70710678118654752f));
    out[i] = (__bf16)(ge * gt);
}

// ---------------------------------------------------------------------------
// Launch
// ---------------------------------------------------------------------------
extern "C" void kernel_launch(void* const* d_in, const int* in_sizes, int n_in,
                              void* d_out, int out_size, void* d_ws, size_t ws_size,
                              hipStream_t stream) {
    const float* hidden = (const float*)d_in[0];
    const int*   amask  = (const int*)d_in[1];
    const float* ln1_w  = (const float*)d_in[2];
    const float* ln1_b  = (const float*)d_in[3];
    const float* wqkv_w = (const float*)d_in[4];
    const float* wqkv_b = (const float*)d_in[5];
    const float* wo_w   = (const float*)d_in[6];
    const float* ln2_w  = (const float*)d_in[7];
    const float* ln2_b  = (const float*)d_in[8];
    const float* wi_w   = (const float*)d_in[9];
    const float* womlp_w= (const float*)d_in[10];
    float* out = (float*)d_out;

    // Workspace layout (bytes, 256-aligned)
    char* ws = (char*)d_ws;
    size_t off = 0;
    auto alloc = [&](size_t bytes) { void* p = ws + off; off += (bytes + 255) & ~(size_t)255; return p; };

    __bf16* xn_bf    = (__bf16*)alloc((size_t)MT_TOT * D_ * 2);          // 8 MB
    float*  qkv_f    = (float*) alloc((size_t)MT_TOT * 3 * D_ * 4);      // 48 MB
    __bf16* q_bf     = (__bf16*)alloc((size_t)MT_TOT * D_ * 2);          // 8 MB
    __bf16* k_bf     = (__bf16*)alloc((size_t)MT_TOT * D_ * 2);          // 8 MB
    __bf16* vT_bf    = (__bf16*)alloc((size_t)MT_TOT * D_ * 2);          // 8 MB
    __bf16* ao_bf    = (__bf16*)alloc((size_t)MT_TOT * D_ * 2);          // 8 MB
    float*  x2_f     = (float*) alloc((size_t)MT_TOT * D_ * 4);          // 16 MB
    __bf16* h_bf     = (__bf16*)alloc((size_t)MT_TOT * D_ * 2);          // 8 MB
    float*  g_f      = (float*) alloc((size_t)MT_TOT * 2 * FF_ * 4);     // 86 MB
    __bf16* act_bf   = (__bf16*)alloc((size_t)MT_TOT * FF_ * 2);         // 21 MB
    __bf16* wqkv_bf  = (__bf16*)alloc((size_t)3 * D_ * D_ * 2);          // 6 MB
    __bf16* wo_bf    = (__bf16*)alloc((size_t)D_ * D_ * 2);              // 2 MB
    __bf16* wi_bf    = (__bf16*)alloc((size_t)2 * FF_ * D_ * 2);         // 10.7 MB
    __bf16* womlp_bf = (__bf16*)alloc((size_t)D_ * FF_ * 2);             // 5.4 MB

    // Pack weights to bf16
    {
        int n;
        n = 3 * D_ * D_;  pack_bf16_kernel<<<(n + 255) / 256, 256, 0, stream>>>(wqkv_w,  wqkv_bf,  n);
        n = D_ * D_;      pack_bf16_kernel<<<(n + 255) / 256, 256, 0, stream>>>(wo_w,    wo_bf,    n);
        n = 2 * FF_ * D_; pack_bf16_kernel<<<(n + 255) / 256, 256, 0, stream>>>(wi_w,    wi_bf,    n);
        n = D_ * FF_;     pack_bf16_kernel<<<(n + 255) / 256, 256, 0, stream>>>(womlp_w, womlp_bf, n);
    }

    // LN1 -> xn (bf16)
    layernorm_bf16_kernel<<<MT_TOT, 256, 0, stream>>>(hidden, ln1_w, ln1_b, xn_bf, D_);

    // QKV GEMM: [4096,1024] @ [3072,1024]^T + bias -> f32
    gemm_bf16_kernel<<<dim3(3 * D_ / GNT, MT_TOT / GMT), 256, 0, stream>>>(
        xn_bf, wqkv_bf, wqkv_b, nullptr, qkv_f, MT_TOT, 3 * D_, D_);

    // RoPE + pack q,k,v
    rope_pack_kernel<<<MT_TOT, 512, 0, stream>>>(qkv_f, q_bf, k_bf, vT_bf);

    // Flash attention
    flash_attn_kernel<<<dim3(M_ / 128, H_, B_), 256, 0, stream>>>(
        q_bf, k_bf, vT_bf, amask, ao_bf);

    // Output projection + residual: x2 = hidden + ao @ wo^T
    gemm_bf16_kernel<<<dim3(D_ / GNT, MT_TOT / GMT), 256, 0, stream>>>(
        ao_bf, wo_bf, nullptr, hidden, x2_f, MT_TOT, D_, D_);

    // LN2 -> h (bf16)
    layernorm_bf16_kernel<<<MT_TOT, 256, 0, stream>>>(x2_f, ln2_w, ln2_b, h_bf, D_);

    // MLP in: g = h @ wi^T  ([4096,1024] @ [5248,1024]^T)
    gemm_bf16_kernel<<<dim3(2 * FF_ / GNT, MT_TOT / GMT), 256, 0, stream>>>(
        h_bf, wi_bf, nullptr, nullptr, g_f, MT_TOT, 2 * FF_, D_);

    // GEGLU -> act (bf16)
    {
        size_t n = (size_t)MT_TOT * FF_;
        geglu_kernel<<<(unsigned)((n + 255) / 256), 256, 0, stream>>>(g_f, act_bf);
    }

    // MLP out + residual: out = x2 + act @ womlp^T  ([4096,2624] @ [1024,2624]^T)
    gemm_bf16_kernel<<<dim3(D_ / GNT, MT_TOT / GMT), 256, 0, stream>>>(
        act_bf, womlp_bf, nullptr, x2_f, out, MT_TOT, D_, FF_);
}
